// EvoformerBlock_22943715295235
// MI455X (gfx1250) — compile-verified, matrix operand order checked by reading, not run on hardware
//
#include <hip/hip_runtime.h>

// ---------------------------------------------------------------------------
// Evoformer block for MI455X (gfx1250, wave32, WMMA bf16 16x16x32).
// All dense contractions run on v_wmma_f32_16x16x32_bf16; attention is fused
// (flash-style online softmax, LDS P-tile transpose so P@V is also WMMA).
// This revision: no strided global gathers in any WMMA K-loop — everything is
// contiguous b128 loads or LDS-staged tiles; OPM a^T produced by epilogue
// scatter; triangle-mult operands permuted to channel-major.
// ---------------------------------------------------------------------------

typedef __attribute__((ext_vector_type(16))) __bf16 bf16x16;
typedef __attribute__((ext_vector_type(8)))  __bf16 bf16x8;
typedef __attribute__((ext_vector_type(8)))  float  f32x8;
typedef __attribute__((ext_vector_type(4)))  float  f32x4;

#define NWAVE 8          // waves per block (8 * wave32 = 256 threads)
#define NTHR  256

__device__ __forceinline__ f32x8 wmma_bf(bf16x16 a, bf16x16 b, f32x8 c) {
  return __builtin_amdgcn_wmma_f32_16x16x32_bf16(false, a, false, b, (short)0, c, false, false);
}
__device__ __forceinline__ int kpat(int kbase, int e) {
  return (e < 8) ? (kbase + e) : (16 + kbase + (e - 8));
}
__device__ __forceinline__ float sigm(float x) { return 1.0f / (1.0f + __expf(-x)); }
__device__ __forceinline__ bf16x8 cvt8(f32x4 a, f32x4 b) {
  bf16x8 r;
#pragma unroll
  for (int i = 0; i < 4; i++) { r[i] = (__bf16)a[i]; r[4 + i] = (__bf16)b[i]; }
  return r;
}
__device__ __forceinline__ bf16x16 cat8(bf16x8 a, bf16x8 b) {
  return __builtin_shufflevector(a, b, 0,1,2,3,4,5,6,7,8,9,10,11,12,13,14,15);
}
// 16-element WMMA fragment from a contiguous 32-float row segment:
// elements 0..7 = p[kbase..kbase+7], 8..15 = p[16+kbase .. 16+kbase+7].
__device__ __forceinline__ bf16x16 frag_row(const float* __restrict__ p, int kbase) {
  const f32x4* p4 = (const f32x4*)(p + kbase);
  f32x4 a0 = p4[0], a1 = p4[1], a2 = p4[4], a3 = p4[5];
  return cat8(cvt8(a0, a1), cvt8(a2, a3));
}
__device__ __forceinline__ bf16x16 frag_row_scaled(const float* __restrict__ p, int kbase, float s) {
  const f32x4* p4 = (const f32x4*)(p + kbase);
  f32x4 a0 = p4[0] * s, a1 = p4[1] * s, a2 = p4[4] * s, a3 = p4[5] * s;
  return cat8(cvt8(a0, a1), cvt8(a2, a3));
}

// ---------------------------------------------------------------------------
// Generic GEMM: Y[M,N] = act(A[M,K] @ B[K,N] + bias) (*gate)(*rowscale)(+res)
// One 16x16 output tile per wave; K consumed 32 at a time. B tile staged
// through a wave-private LDS buffer with coalesced b128 loads.
// store_mode 1 = OPM scatter: M-row (s*256+i), col c -> Y[(i*32+c)*128 + s].
// ---------------------------------------------------------------------------
__global__ void gemm_wmma(const float* __restrict__ A, int lda,
                          const float* __restrict__ B, int ldb,
                          float* __restrict__ Y, int ldy,
                          int M, int N, int K,
                          const float* __restrict__ bias, int relu,
                          const float* __restrict__ gate,
                          const float* __restrict__ rowscale,
                          const float* __restrict__ res,
                          int store_mode) {
  __shared__ __bf16 btile[NWAVE][32 * 16];
  int lane = threadIdx.x & 31, wave = threadIdx.x >> 5;
  long wgl = (long)blockIdx.x * NWAVE + wave;
  int ntn = N >> 4;
  long ntiles = (long)(M >> 4) * ntn;
  if (wgl >= ntiles) return;
  int tn = (int)(wgl % ntn);
  int tm = (int)(wgl / ntn);
  int lr = lane & 15, hi = lane >> 4, kbase = hi * 8;
  int row_a = tm * 16 + lr;
  int col_b = tn * 16 + lr;
  const float* arow = A + (long)row_a * lda;     // one 64-bit base, then int offsets
  const float* brow = B + (long)lane * ldb + tn * 16;
  __bf16* bt = &btile[wave][0];
  f32x8 acc = {0.f,0.f,0.f,0.f,0.f,0.f,0.f,0.f};
  for (int k0 = 0; k0 < K; k0 += 32) {
    // --- stage B tile [32k x 16n] into LDS (coalesced b128) ---
    {
      const f32x4* bp4 = (const f32x4*)(brow + k0 * ldb);
      f32x4 b0 = bp4[0], b1 = bp4[1], b2 = bp4[2], b3 = bp4[3];
      bf16x8* dst = (bf16x8*)(bt + lane * 16);
      dst[0] = cvt8(b0, b1);
      dst[1] = cvt8(b2, b3);
    }
    // --- A fragment (contiguous b128) ---
    bf16x16 af = frag_row(arow + k0, kbase);
    if (k0 + 32 < K) __builtin_prefetch(arow + k0 + 32, 0, 1);
    // --- B fragment from LDS (per-lane column lr) ---
    bf16x16 bf;
#pragma unroll
    for (int e = 0; e < 16; e++) bf[e] = bt[kpat(kbase, e) * 16 + lr];
    acc = wmma_bf(af, bf, acc);
  }
#pragma unroll
  for (int r = 0; r < 8; r++) {
    int row = tm * 16 + hi * 8 + r;
    int col = col_b;
    int idx = row * ldy + col;
    float t = acc[r];
    if (bias)     t += bias[col];
    if (relu)     t = fmaxf(t, 0.f);
    if (gate)     t *= sigm(gate[idx]);
    if (rowscale) t *= rowscale[row];
    if (res)      t += res[idx];
    int oidx = (store_mode == 0) ? idx : (((row & 255) * 32 + col) * 128 + (row >> 8));
    Y[oidx] = t;
  }
}

// ---------------------------------------------------------------------------
// LayerNorm over last dim C (one row per wave).
// ---------------------------------------------------------------------------
__global__ void layernorm_k(const float* __restrict__ x, float* __restrict__ y,
                            const float* __restrict__ g, const float* __restrict__ b,
                            int C, int rows) {
  int lane = threadIdx.x & 31, wave = threadIdx.x >> 5;
  int row = blockIdx.x * NWAVE + wave;
  if (row >= rows) return;
  const float* xr = x + (long)row * C;
  float s0 = 0.f, s1 = 0.f;
  for (int c = lane; c < C; c += 32) { float v = xr[c]; s0 += v; s1 += v * v; }
  for (int m = 16; m >= 1; m >>= 1) { s0 += __shfl_xor(s0, m, 32); s1 += __shfl_xor(s1, m, 32); }
  float mean = s0 / C, var = s1 / C - mean * mean;
  float inv = rsqrtf(var + 1e-5f);
  float* yr = y + (long)row * C;
  for (int c = lane; c < C; c += 32) yr[c] = (xr[c] - mean) * inv * g[c] + b[c];
}

// Small-N projection (attention pair-bias heads, N = 8 or 4).
__global__ void smallproj(const float* __restrict__ X, const float* __restrict__ W,
                          float* __restrict__ Y, int C, int N, int rows) {
  long tid = (long)blockIdx.x * NTHR + threadIdx.x;
  if (tid >= (long)rows * N) return;
  int row = (int)(tid / N), n = (int)(tid % N);
  const float* xr = X + (long)row * C;
  float acc = 0.f;
  for (int c = 0; c < C; c++) acc += xr[c] * W[c * N + n];
  Y[tid] = acc;
}

// O *= sigmoid(G) elementwise.
__global__ void gate_mul(float* __restrict__ O, const float* __restrict__ G, int n) {
  int i = blockIdx.x * NTHR + threadIdx.x;
  if (i < n) O[i] *= sigm(G[i]);
}

// out = rowmask * sigmoid(G) * P  (triangle-mult input gating), C channels/row.
__global__ void gate_mask_mul(float* __restrict__ out, const float* __restrict__ G,
                              const float* __restrict__ P, const float* __restrict__ rowmask,
                              int C, int n) {
  int i = blockIdx.x * NTHR + threadIdx.x;
  if (i < n) out[i] = rowmask[i / C] * sigm(G[i]) * P[i];
}

// Permute [row(p,q), C] -> channel-major [c][p][q]; incoming variant reads the
// (q,p) element so the k-transpose is absorbed here.
__global__ void reorder_tri(const float* __restrict__ in, float* __restrict__ out,
                            int R, int C, int outgoing) {
  int idx = blockIdx.x * NTHR + threadIdx.x;   // writes coalesced (q fastest)
  int RR = R * R;
  int c = idx / RR, rem = idx % RR;
  int p = rem / R, q = rem % R;
  int src = outgoing ? ((p * R + q) * C + c) : ((q * R + p) * C + c);
  out[idx] = in[src];
}

// norm[i,j] = sum_s mask[s,i]*mask[s,j] + 1e-3
__global__ void opm_norm(const float* __restrict__ mask, float* __restrict__ norm, int S, int R) {
  int idx = blockIdx.x * NTHR + threadIdx.x;
  if (idx >= R * R) return;
  int i = idx / R, j = idx % R;
  float acc = 0.f;
  for (int s = 0; s < S; s++) acc += mask[s * R + i] * mask[s * R + j];
  norm[idx] = acc + 1e-3f;
}

// ---------------------------------------------------------------------------
// Fused attention (all four variants): one wave = one 16-query tile of one
// head of one batch slice. Q@K^T on WMMA, online softmax with half-wave
// shfl reductions, P and V staged through LDS so P@V is WMMA too.
// ---------------------------------------------------------------------------
__global__ void attn_wmma(const float* __restrict__ Qp, const float* __restrict__ Kp,
                          const float* __restrict__ Vp, float* __restrict__ Op,
                          int b_stride, int row_stride,
                          int nq, int nk, int nheads,
                          const float* __restrict__ biasp, int bias_swap, int bias_H,
                          const float* __restrict__ maskp, int mask_b, int mask_k,
                          float scale) {
  __shared__ __bf16 pshare[NWAVE][16 * 32];
  __shared__ __bf16 vshare[NWAVE][32 * 32];
  int lane = threadIdx.x & 31, wave = threadIdx.x >> 5;
  int wg = blockIdx.x * NWAVE + wave;          // grid is exact by construction
  int qt_n = nq >> 4;
  int qt = wg % qt_n;
  int h  = (wg / qt_n) % nheads;
  int bI = wg / (qt_n * nheads);
  int lr = lane & 15, hi = lane >> 4, kbase = hi * 8;
  int boff = bI * b_stride + h * 32;
  const float* qbase = Qp + boff;
  const float* kbase_p = Kp + boff;
  const float* vbase = Vp + boff;

  // Q fragment (rows i, K = 32 channels), softmax scale folded in.
  bf16x16 aq = frag_row_scaled(qbase + (qt * 16 + lr) * row_stride, kbase, scale);

  f32x8 z8 = {0.f,0.f,0.f,0.f,0.f,0.f,0.f,0.f};
  f32x8 acc0 = z8, acc1 = z8;
  float rmax[8], rsum[8];
#pragma unroll
  for (int r = 0; r < 8; r++) { rmax[r] = -1e30f; rsum[r] = 0.f; }
  __bf16* pl = &pshare[wave][0];
  __bf16* vt = &vshare[wave][0];

  int nchunks = nk >> 5;
  for (int ch = 0; ch < nchunks; ch++) {
    int j0 = ch * 32;
    // --- stage V tile [32k x 32c] into LDS (coalesced b128) ---
    {
      const f32x4* v4 = (const f32x4*)(vbase + (j0 + lane) * row_stride);
      bf16x8* dst = (bf16x8*)(vt + lane * 32);
      dst[0] = cvt8(v4[0], v4[1]);
      dst[1] = cvt8(v4[2], v4[3]);
      dst[2] = cvt8(v4[4], v4[5]);
      dst[3] = cvt8(v4[6], v4[7]);
    }
    float sv[2][8];
#pragma unroll
    for (int t = 0; t < 2; t++) {
      int kj = j0 + t * 16 + lr;
      bf16x16 bk = frag_row(kbase_p + kj * row_stride, kbase);
      f32x8 s = wmma_bf(aq, bk, z8);
      float madd = 0.f;
      if (maskp) madd = (maskp[bI * mask_b + kj * mask_k] - 1.0f) * 1e9f;
#pragma unroll
      for (int r = 0; r < 8; r++) {
        float tv = s[r] + madd;
        if (biasp) {
          int qi = qt * 16 + hi * 8 + r;
          int bidx = bias_swap ? ((kj * nk + qi) * bias_H + h)
                               : ((qi * nk + kj) * bias_H + h);
          tv += biasp[bidx];
        }
        sv[t][r] = tv;
      }
    }
    // online softmax (rows live across 16-lane half groups)
    float cmax[8];
#pragma unroll
    for (int r = 0; r < 8; r++) cmax[r] = fmaxf(sv[0][r], sv[1][r]);
    for (int m = 1; m <= 8; m <<= 1)
#pragma unroll
      for (int r = 0; r < 8; r++) cmax[r] = fmaxf(cmax[r], __shfl_xor(cmax[r], m, 32));
    float corr[8], p0[8], p1[8], csum[8];
#pragma unroll
    for (int r = 0; r < 8; r++) {
      float nm = fmaxf(rmax[r], cmax[r]);
      corr[r] = __expf(rmax[r] - nm);
      rmax[r] = nm;
      p0[r] = __expf(sv[0][r] - nm);
      p1[r] = __expf(sv[1][r] - nm);
      csum[r] = p0[r] + p1[r];
    }
    for (int m = 1; m <= 8; m <<= 1)
#pragma unroll
      for (int r = 0; r < 8; r++) csum[r] += __shfl_xor(csum[r], m, 32);
#pragma unroll
    for (int r = 0; r < 8; r++) {
      rsum[r] = rsum[r] * corr[r] + csum[r];
      acc0[r] *= corr[r];
      acc1[r] *= corr[r];
    }
    // stage P tile (16 rows x 32 keys) in LDS as bf16, D-layout writes
#pragma unroll
    for (int r = 0; r < 8; r++) {
      pl[(hi * 8 + r) * 32 + lr]      = (__bf16)p0[r];
      pl[(hi * 8 + r) * 32 + 16 + lr] = (__bf16)p1[r];
    }
    __syncthreads();
    bf16x8 plo = *(const bf16x8*)(pl + lr * 32 + kbase);
    bf16x8 phi = *(const bf16x8*)(pl + lr * 32 + 16 + kbase);
    bf16x16 pa = cat8(plo, phi);
    // V fragments gathered from the LDS-staged tile
    bf16x16 bv0, bv1;
#pragma unroll
    for (int e = 0; e < 16; e++) {
      int kk = kpat(kbase, e) * 32;
      bv0[e] = vt[kk + lr];
      bv1[e] = vt[kk + 16 + lr];
    }
    __syncthreads();
    acc0 = wmma_bf(pa, bv0, acc0);
    acc1 = wmma_bf(pa, bv1, acc1);
  }
#pragma unroll
  for (int r = 0; r < 8; r++) {
    int qi = qt * 16 + hi * 8 + r;
    float inv = 1.0f / rsum[r];
    float* op = Op + boff + qi * row_stride;
    op[lr]      = acc0[r] * inv;
    op[16 + lr] = acc1[r] * inv;
  }
}

// ---------------------------------------------------------------------------
// Triangle multiplication on channel-major operands: per channel c,
// X[i,j,c] = sum_k Ac[c][i][k] * Bc[c][j][k]  (both fragments contiguous).
// ---------------------------------------------------------------------------
__global__ void trimul_cm(const float* __restrict__ Ac, const float* __restrict__ Bc,
                          float* __restrict__ X, int R, int C) {
  int lane = threadIdx.x & 31, wave = threadIdx.x >> 5;
  int wg = blockIdx.x * NWAVE + wave;
  int tiles = (R >> 4) * (R >> 4);
  int c  = wg / tiles;
  int tt = wg % tiles;
  int ti = tt / (R >> 4), tj = tt % (R >> 4);
  int lr = lane & 15, hi = lane >> 4, kbase = hi * 8;
  int ia = ti * 16 + lr, jb = tj * 16 + lr;
  const float* arow = Ac + c * R * R + ia * R;
  const float* brow = Bc + c * R * R + jb * R;
  f32x8 acc = {0.f,0.f,0.f,0.f,0.f,0.f,0.f,0.f};
  for (int k0 = 0; k0 < R; k0 += 32) {
    bf16x16 af = frag_row(arow + k0, kbase);
    bf16x16 bf = frag_row(brow + k0, kbase);
    acc = wmma_bf(af, bf, acc);
  }
#pragma unroll
  for (int r = 0; r < 8; r++) {
    int i = ti * 16 + hi * 8 + r;
    X[(i * R + jb) * C + c] = acc[r];
  }
}

// ---------------------------------------------------------------------------
// OPM contraction: for an i-chunk of 32, outer_chunk[(il*32+c), (j*32+d)]
// contracted with wo[(c*32+d), e]; Wo tile LDS-staged; epilogue /norm, += z.
// ---------------------------------------------------------------------------
__global__ void opm_contract(const float* __restrict__ OC, const float* __restrict__ Wo,
                             const float* __restrict__ bo, const float* __restrict__ norm,
                             float* __restrict__ Z, int i0) {
  __shared__ __bf16 wtile[NWAVE][32 * 16];
  int lane = threadIdx.x & 31, wave = threadIdx.x >> 5;
  int wg = blockIdx.x * NWAVE + wave;          // M=8192 pairs, N=128 -> 4096 waves
  int tn = wg % 8;
  int tm = wg / 8;
  int lr = lane & 15, hi = lane >> 4, kbase = hi * 8;
  int pair_a = tm * 16 + lr;                   // (il*256 + j)
  int il_a = pair_a >> 8, j_a = pair_a & 255;
  int n = tn * 16 + lr;
  const float* oc_row0 = OC + il_a * 32 * 8192 + j_a * 32;  // row for c=0
  const float* wrow = Wo + lane * 128 + tn * 16;
  __bf16* wt = &wtile[wave][0];
  f32x8 acc = {0.f,0.f,0.f,0.f,0.f,0.f,0.f,0.f};
  for (int k0 = 0; k0 < 1024; k0 += 32) {
    {
      const f32x4* wp4 = (const f32x4*)(wrow + k0 * 128);
      f32x4 w0 = wp4[0], w1 = wp4[1], w2 = wp4[2], w3 = wp4[3];
      bf16x8* dst = (bf16x8*)(wt + lane * 16);
      dst[0] = cvt8(w0, w1);
      dst[1] = cvt8(w2, w3);
    }
    int cc = k0 >> 5;                          // K step of 32 == one c
    bf16x16 af = frag_row(oc_row0 + cc * 8192, kbase);   // contiguous in d
    bf16x16 bf;
#pragma unroll
    for (int e = 0; e < 16; e++) bf[e] = wt[kpat(kbase, e) * 16 + lr];
    acc = wmma_bf(af, bf, acc);
  }
#pragma unroll
  for (int r = 0; r < 8; r++) {
    int pair = tm * 16 + hi * 8 + r;
    int il = pair >> 8, j = pair & 255;
    int gi = i0 + il;
    float t = (acc[r] + bo[n]) / norm[gi * 256 + j];
    int zi = (gi * 256 + j) * 128 + n;
    Z[zi] += t;
  }
}

// ---------------------------------------------------------------------------
// Host orchestration
// ---------------------------------------------------------------------------
static inline void gemm(hipStream_t st, const float* A, int lda,
                        const float* B, int ldb, float* Y, int ldy,
                        int M, int N, int K, const float* bias, int relu,
                        const float* gate, const float* rowscale, const float* res,
                        int store_mode = 0) {
  long tiles = (long)(M / 16) * (N / 16);
  int blocks = (int)((tiles + NWAVE - 1) / NWAVE);
  gemm_wmma<<<blocks, NTHR, 0, st>>>(A, lda, B, ldb, Y, ldy, M, N, K, bias, relu, gate, rowscale, res, store_mode);
}
static inline void lnorm(hipStream_t st, const float* x, float* y, const float* g,
                         const float* b, int C, int rows) {
  layernorm_k<<<rows / NWAVE, NTHR, 0, st>>>(x, y, g, b, C, rows);
}

// Param indices: jax pytree (alphabetical) flattening of the params dict.
enum {
  P_ca_bg=0,P_ca_bo,P_ca_ln_b,P_ca_ln_g,P_ca_wg,P_ca_wk,P_ca_wo,P_ca_wq,P_ca_wv,
  P_mt_b1,P_mt_b2,P_mt_ln_b,P_mt_ln_g,P_mt_w1,P_mt_w2,
  P_opm_ba,P_opm_bb,P_opm_bo,P_opm_ln_b,P_opm_ln_g,P_opm_wa,P_opm_wb,P_opm_wo,
  P_pt_b1,P_pt_b2,P_pt_ln_b,P_pt_ln_g,P_pt_w1,P_pt_w2,
  P_ra_bg,P_ra_bo,P_ra_ln_b,P_ra_ln_g,P_ra_lnz_b,P_ra_lnz_g,P_ra_wbias,P_ra_wg,P_ra_wk,P_ra_wo,P_ra_wq,P_ra_wv,
  P_tae_bg,P_tae_bo,P_tae_ln_b,P_tae_ln_g,P_tae_wbias,P_tae_wg,P_tae_wk,P_tae_wo,P_tae_wq,P_tae_wv,
  P_tas_bg,P_tas_bo,P_tas_ln_b,P_tas_ln_g,P_tas_wbias,P_tas_wg,P_tas_wk,P_tas_wo,P_tas_wq,P_tas_wv,
  P_tmi_bag,P_tmi_bap,P_tmi_bbg,P_tmi_bbp,P_tmi_bgate,P_tmi_bo,P_tmi_ln_b,P_tmi_ln_g,P_tmi_lnx_b,P_tmi_lnx_g,
  P_tmi_wag,P_tmi_wap,P_tmi_wbg,P_tmi_wbp,P_tmi_wgate,P_tmi_wo,
  P_tmo_bag,P_tmo_bap,P_tmo_bbg,P_tmo_bbp,P_tmo_bgate,P_tmo_bo,P_tmo_ln_b,P_tmo_ln_g,P_tmo_lnx_b,P_tmo_lnx_g,
  P_tmo_wag,P_tmo_wap,P_tmo_wbg,P_tmo_wbp,P_tmo_wgate,P_tmo_wo
};

extern "C" void kernel_launch(void* const* d_in, const int* in_sizes, int n_in,
                              void* d_out, int out_size, void* d_ws, size_t ws_size,
                              hipStream_t stream) {
  const int S = 128, R = 256, CM = 256, CZ = 128;
  const int SR = S * R;                // 32768
  const int RR = R * R;                // 65536
  const int SRCM = SR * CM;            // 8388608
  const int RRCZ = RR * CZ;            // 8388608
  const float SC = 0.17677669529663687f;  // 1/sqrt(32)

  const float* m_in  = (const float*)d_in[0];
  const float* z_in  = (const float*)d_in[1];
  const float* mmask = (const float*)d_in[2];
  const float* pmask = (const float*)d_in[3];
  auto P = [&](int i) { return (const float*)d_in[4 + i]; };

  // workspace layout: 32MB (8M-float) slots
  const long SLOT = 8388608;
  float* ws   = (float*)d_ws;
  float* MCUR = ws + 0 * SLOT;
  float* ZCUR = ws + 1 * SLOT;
  float* LNB  = ws + 2 * SLOT;
  float* QB   = ws + 3 * SLOT;
  float* KB   = ws + 4 * SLOT;
  float* VB   = ws + 5 * SLOT;
  float* GB   = ws + 6 * SLOT;
  float* OB   = ws + 7 * SLOT;
  float* T1   = ws + 8 * SLOT;   // up to 128MB (slots 8..11)
  float* T2   = ws + 12 * SLOT;
  float* XB   = ws + 13 * SLOT;
  float* BIASB = ws + 14 * SLOT;
  float* NORMB = BIASB + 524288;

  hipMemcpyAsync(MCUR, m_in, (long)SRCM * 4, hipMemcpyDeviceToDevice, stream);
  hipMemcpyAsync(ZCUR, z_in, (long)RRCZ * 4, hipMemcpyDeviceToDevice, stream);

  // ---- 1. MSA row attention with pair bias -------------------------------
  lnorm(stream, MCUR, LNB, P(P_ra_ln_g), P(P_ra_ln_b), CM, SR);
  lnorm(stream, ZCUR, XB, P(P_ra_lnz_g), P(P_ra_lnz_b), CZ, RR);
  gemm(stream, LNB, CM, P(P_ra_wq), CM, QB, CM, SR, CM, CM, 0, 0, 0, 0, 0);
  gemm(stream, LNB, CM, P(P_ra_wk), CM, KB, CM, SR, CM, CM, 0, 0, 0, 0, 0);
  gemm(stream, LNB, CM, P(P_ra_wv), CM, VB, CM, SR, CM, CM, 0, 0, 0, 0, 0);
  gemm(stream, LNB, CM, P(P_ra_wg), CM, GB, CM, SR, CM, CM, P(P_ra_bg), 0, 0, 0, 0);
  smallproj<<<2048, NTHR, 0, stream>>>(XB, P(P_ra_wbias), BIASB, CZ, 8, RR);
  attn_wmma<<<2048, NTHR, 0, stream>>>(QB, KB, VB, OB, R * CM, CM, R, R, 8,
                                       BIASB, 0, 8, mmask, R, 1, SC);
  gate_mul<<<32768, NTHR, 0, stream>>>(OB, GB, SRCM);
  gemm(stream, OB, CM, P(P_ra_wo), CM, MCUR, CM, SR, CM, CM, P(P_ra_bo), 0, 0, 0, MCUR);

  // ---- 2. MSA column attention ------------------------------------------
  lnorm(stream, MCUR, LNB, P(P_ca_ln_g), P(P_ca_ln_b), CM, SR);
  gemm(stream, LNB, CM, P(P_ca_wq), CM, QB, CM, SR, CM, CM, 0, 0, 0, 0, 0);
  gemm(stream, LNB, CM, P(P_ca_wk), CM, KB, CM, SR, CM, CM, 0, 0, 0, 0, 0);
  gemm(stream, LNB, CM, P(P_ca_wv), CM, VB, CM, SR, CM, CM, 0, 0, 0, 0, 0);
  gemm(stream, LNB, CM, P(P_ca_wg), CM, GB, CM, SR, CM, CM, P(P_ca_bg), 0, 0, 0, 0);
  attn_wmma<<<2048, NTHR, 0, stream>>>(QB, KB, VB, OB, CM, R * CM, S, S, 8,
                                       (const float*)0, 0, 0, mmask, 1, R, SC);
  gate_mul<<<32768, NTHR, 0, stream>>>(OB, GB, SRCM);
  gemm(stream, OB, CM, P(P_ca_wo), CM, MCUR, CM, SR, CM, CM, P(P_ca_bo), 0, 0, 0, MCUR);

  // ---- 3. MSA transition -------------------------------------------------
  lnorm(stream, MCUR, LNB, P(P_mt_ln_g), P(P_mt_ln_b), CM, SR);
  gemm(stream, LNB, CM, P(P_mt_w1), 1024, T1, 1024, SR, 1024, CM, P(P_mt_b1), 1, 0, 0, 0);
  gemm(stream, T1, 1024, P(P_mt_w2), CM, MCUR, CM, SR, CM, 1024, P(P_mt_b2), 0, 0, mmask, MCUR);

  // ---- 4. Outer product mean --------------------------------------------
  lnorm(stream, MCUR, LNB, P(P_opm_ln_g), P(P_opm_ln_b), CM, SR);
  // a projected straight into transposed [(i*32+c), s] layout via scatter epilogue
  gemm(stream, LNB, CM, P(P_opm_wa), 32, QB, 32, SR, 32, CM, P(P_opm_ba), 0, 0, mmask, 0, 1);
  gemm(stream, LNB, CM, P(P_opm_wb), 32, KB, 32, SR, 32, CM, P(P_opm_bb), 0, 0, mmask, 0, 0);
  opm_norm<<<256, NTHR, 0, stream>>>(mmask, NORMB, S, R);
  for (int ic = 0; ic < 8; ic++) {
    int i0 = ic * 32;
    // outer chunk [1024, 8192] = aT(chunk rows)[1024,128] @ b[128,8192]
    gemm(stream, QB + i0 * 32 * 128, 128, KB, R * 32, T1, R * 32, 1024, R * 32, S, 0, 0, 0, 0, 0);
    opm_contract<<<512, NTHR, 0, stream>>>(T1, P(P_opm_wo), P(P_opm_bo), NORMB, ZCUR, i0);
  }

  // ---- 5/6. Triangle multiplication outgoing / incoming ------------------
  const int TMW[2][12] = {
    {P_tmo_ln_g,P_tmo_ln_b,P_tmo_wag,P_tmo_bag,P_tmo_wap,P_tmo_bap,P_tmo_wbg,P_tmo_bbg,P_tmo_wbp,P_tmo_bbp,P_tmo_wgate,P_tmo_bgate},
    {P_tmi_ln_g,P_tmi_ln_b,P_tmi_wag,P_tmi_bag,P_tmi_wap,P_tmi_bap,P_tmi_wbg,P_tmi_bbg,P_tmi_wbp,P_tmi_bbp,P_tmi_wgate,P_tmi_bgate}};
  const int TMX[2][4] = {
    {P_tmo_lnx_g,P_tmo_lnx_b,P_tmo_wo,P_tmo_bo},
    {P_tmi_lnx_g,P_tmi_lnx_b,P_tmi_wo,P_tmi_bo}};
  for (int v = 0; v < 2; v++) {
    int outg = (v == 0) ? 1 : 0;
    lnorm(stream, ZCUR, LNB, P(TMW[v][0]), P(TMW[v][1]), CZ, RR);
    gemm(stream, LNB, CZ, P(TMW[v][10]), CZ, GB, CZ, RR, CZ, CZ, P(TMW[v][11]), 0, 0, 0, 0);
    gemm(stream, LNB, CZ, P(TMW[v][2]), CZ, QB, CZ, RR, CZ, CZ, P(TMW[v][3]), 0, 0, 0, 0);
    gemm(stream, LNB, CZ, P(TMW[v][4]), CZ, KB, CZ, RR, CZ, CZ, P(TMW[v][5]), 0, 0, 0, 0);
    gate_mask_mul<<<32768, NTHR, 0, stream>>>(VB, QB, KB, pmask, CZ, RRCZ);
    reorder_tri<<<32768, NTHR, 0, stream>>>(VB, T1, R, CZ, outg);
    gemm(stream, LNB, CZ, P(TMW[v][6]), CZ, QB, CZ, RR, CZ, CZ, P(TMW[v][7]), 0, 0, 0, 0);
    gemm(stream, LNB, CZ, P(TMW[v][8]), CZ, KB, CZ, RR, CZ, CZ, P(TMW[v][9]), 0, 0, 0, 0);
    gate_mask_mul<<<32768, NTHR, 0, stream>>>(VB, QB, KB, pmask, CZ, RRCZ);
    reorder_tri<<<32768, NTHR, 0, stream>>>(VB, T2, R, CZ, outg);
    trimul_cm<<<4096, NTHR, 0, stream>>>(T1, T2, XB, R, CZ);
    lnorm(stream, XB, OB, P(TMX[v][0]), P(TMX[v][1]), CZ, RR);
    gemm(stream, OB, CZ, P(TMX[v][2]), CZ, ZCUR, CZ, RR, CZ, CZ, P(TMX[v][3]), 0, GB, 0, ZCUR);
  }

  // ---- 7/8. Triangle attention (starting / ending node) ------------------
  const int TA[2][10] = {
    {P_tas_ln_g,P_tas_ln_b,P_tas_wq,P_tas_wk,P_tas_wv,P_tas_wg,P_tas_bg,P_tas_wbias,P_tas_wo,P_tas_bo},
    {P_tae_ln_g,P_tae_ln_b,P_tae_wq,P_tae_wk,P_tae_wv,P_tae_wg,P_tae_bg,P_tae_wbias,P_tae_wo,P_tae_bo}};
  for (int v = 0; v < 2; v++) {
    lnorm(stream, ZCUR, LNB, P(TA[v][0]), P(TA[v][1]), CZ, RR);
    gemm(stream, LNB, CZ, P(TA[v][2]), CZ, QB, CZ, RR, CZ, CZ, 0, 0, 0, 0, 0);
    gemm(stream, LNB, CZ, P(TA[v][3]), CZ, KB, CZ, RR, CZ, CZ, 0, 0, 0, 0, 0);
    gemm(stream, LNB, CZ, P(TA[v][4]), CZ, VB, CZ, RR, CZ, CZ, 0, 0, 0, 0, 0);
    gemm(stream, LNB, CZ, P(TA[v][5]), CZ, GB, CZ, RR, CZ, CZ, P(TA[v][6]), 0, 0, 0, 0);
    smallproj<<<1024, NTHR, 0, stream>>>(LNB, P(TA[v][7]), BIASB, CZ, 4, RR);
    if (v == 0) {  // starting node: batch = i, rows = j, keys = k
      attn_wmma<<<2048, NTHR, 0, stream>>>(QB, KB, VB, OB, R * CZ, CZ, R, R, 4,
                                           BIASB, 0, 4, pmask, R, 1, SC);
    } else {       // ending node: transpose folded into strides + swapped bias/mask
      attn_wmma<<<2048, NTHR, 0, stream>>>(QB, KB, VB, OB, CZ, R * CZ, R, R, 4,
                                           BIASB, 1, 4, pmask, 1, R, SC);
    }
    gate_mul<<<32768, NTHR, 0, stream>>>(OB, GB, RRCZ);
    gemm(stream, OB, CZ, P(TA[v][8]), CZ, ZCUR, CZ, RR, CZ, CZ, P(TA[v][9]), 0, 0, 0, ZCUR);
  }

  // ---- 9. Pair transition ------------------------------------------------
  lnorm(stream, ZCUR, LNB, P(P_pt_ln_g), P(P_pt_ln_b), CZ, RR);
  gemm(stream, LNB, CZ, P(P_pt_w1), 512, T1, 512, RR, 512, CZ, P(P_pt_b1), 1, 0, 0, 0);
  gemm(stream, T1, 512, P(P_pt_w2), CZ, ZCUR, CZ, RR, CZ, 512, P(P_pt_b2), 0, 0, pmask, ZCUR);

  // ---- outputs: (m, z) concatenated -------------------------------------
  hipMemcpyAsync((float*)d_out, MCUR, (long)SRCM * 4, hipMemcpyDeviceToDevice, stream);
  hipMemcpyAsync((float*)d_out + SRCM, ZCUR, (long)RRCZ * 4, hipMemcpyDeviceToDevice, stream);
}
